// GCNBaseline_52948356825196
// MI455X (gfx1250) — matrix operand order repaired and checked by reading, not compiled, not run on hardware
//
#include <hip/hip_runtime.h>
#include <hip/hip_bf16.h>

// ---------------------------------------------------------------------------
// GCN dual-stream encoder + head for MI455X (gfx1250, wave32, WMMA bf16).
// B=64, N=1024, HID=64, EMB=128, NCLS=2.
// ---------------------------------------------------------------------------

#define Bn   64
#define Nn   1024
#define HIDn 64
#define EMBn 128
#define TILE 64
#define Tt   (Nn / TILE)            // 16 tile-rows
#define NPAIRS (Tt * (Tt + 1) / 2)  // 136 upper-triangle tile pairs

// Flip to 0 if the async-LDS inline asm fails to assemble.
#define USE_ASYNC_STAGE 1

typedef __attribute__((ext_vector_type(16))) __bf16 v16bf;
typedef __attribute__((ext_vector_type(8)))  float  v8f;

union Frag16 { v16bf v; unsigned u[8]; uint4 q[2]; };
union Frag32 { v8f  v; float   f[8]; };

// fp32 -> bf16 with round-to-nearest (tie-up): 2 VALU ops
__device__ __forceinline__ unsigned short f2bf(float f) {
  unsigned u = __builtin_bit_cast(unsigned, f) + 0x8000u;
  return (unsigned short)(u >> 16);
}

// Packed fp32x2 -> bf16x2 (low = a, high = b): 2 adds + 1 v_perm_b32
__device__ __forceinline__ unsigned pk2bf(float a, float b) {
  unsigned ua = __builtin_bit_cast(unsigned, a) + 0x8000u;
  unsigned ub = __builtin_bit_cast(unsigned, b) + 0x8000u;
#if __has_builtin(__builtin_amdgcn_perm)
  // dst bytes: [0]=a.b2 [1]=a.b3 [2]=b.b2 [3]=b.b3 ; sel<4 -> src1, sel>=4 -> src0
  return __builtin_amdgcn_perm(ub, ua, 0x07060302u);
#else
  return (ua >> 16) | (ub & 0xffff0000u);
#endif
}

__device__ __forceinline__ void pair_decode(int p, int& ti, int& tj) {
  int t = 0;
  while (p >= Tt - t) { p -= Tt - t; ++t; }
  ti = t; tj = t + p;
}

// ------------------------------- utility -----------------------------------
__global__ void set_kernel(float* p, int n, float v) {
  int i = blockIdx.x * blockDim.x + threadIdx.x;
  if (i < n) p[i] = v;
}

__global__ void finalize_dinv_kernel(float* p, int n) {
  int i = blockIdx.x * blockDim.x + threadIdx.x;
  if (i < n) { float d = p[i]; p[i] = (d > 0.f) ? rsqrtf(d) : 0.f; }
}

// -------------------- degree accumulation (strict upper) -------------------
__global__ __launch_bounds__(256)
void deg_accum_kernel(const float* __restrict__ adj, float* __restrict__ deg) {
  __shared__ float s[TILE][TILE + 1];
  int ti, tj; pair_decode(blockIdx.x, ti, tj);
  const int b = blockIdx.y;
  const int t = threadIdx.x;
  const int r = t >> 2, q = t & 3;
  const float* A = adj + (size_t)b * Nn * Nn;

  float rs = 0.f;
  #pragma unroll
  for (int m = 0; m < 16; ++m) {
    int jl = q * 16 + m;
    float v = A[(size_t)(ti * TILE + r) * Nn + (tj * TILE + jl)];
    bool upper = (ti < tj) || (r < jl);
    v = upper ? v : 0.f;
    s[r][jl] = v;
    rs += v;
  }
  atomicAdd(&deg[(size_t)b * Nn + ti * TILE + r], rs);
  __syncthreads();
  float cs = 0.f;
  #pragma unroll
  for (int m = 0; m < 16; ++m) cs += s[q * 16 + m][r];
  atomicAdd(&deg[(size_t)b * Nn + tj * TILE + r], cs);
}

// ------------------- build normalized adjacency (bf16) ---------------------
__global__ __launch_bounds__(256)
void build_A_kernel(const float* __restrict__ adj, const float* __restrict__ dinv,
                    unsigned short* __restrict__ Abf) {
  __shared__ float s[TILE][TILE + 1];
  __shared__ float sdi[TILE], sdj[TILE];
  int ti, tj; pair_decode(blockIdx.x, ti, tj);
  const int b = blockIdx.y;
  const int t = threadIdx.x;
  const int r = t >> 2, q = t & 3;
  const float* A = adj + (size_t)b * Nn * Nn;
  unsigned short* O = Abf + (size_t)b * Nn * Nn;

  if (t < TILE)          sdi[t] = dinv[(size_t)b * Nn + ti * TILE + t];
  else if (t < 2 * TILE) sdj[t - TILE] = dinv[(size_t)b * Nn + tj * TILE + (t - TILE)];

  #pragma unroll
  for (int m = 0; m < 16; ++m) {
    int jl = q * 16 + m;
    s[r][jl] = A[(size_t)(ti * TILE + r) * Nn + (tj * TILE + jl)];
  }
  __syncthreads();

  // upper output tile (ti, tj) — packed dword stores (jl pairs)
  #pragma unroll
  for (int m = 0; m < 16; m += 2) {
    int jl = q * 16 + m;
    float r0, r1;
    if (ti == tj) {
      r0 = (r < jl)     ? s[r][jl]     : ((r > jl)     ? s[jl][r]     : 1.0f);
      r1 = (r < jl + 1) ? s[r][jl + 1] : ((r > jl + 1) ? s[jl + 1][r] : 1.0f);
    } else {
      r0 = s[r][jl]; r1 = s[r][jl + 1];
    }
    unsigned pkd = pk2bf(sdi[r] * sdj[jl] * r0, sdi[r] * sdj[jl + 1] * r1);
    *(unsigned*)(O + (size_t)(ti * TILE + r) * Nn + (tj * TILE + jl)) = pkd;
  }
  // mirrored lower output tile (tj, ti)
  if (ti != tj) {
    #pragma unroll
    for (int m = 0; m < 16; m += 2) {
      int jl = q * 16 + m;
      unsigned pkd = pk2bf(sdj[r] * sdi[jl] * s[jl][r],
                           sdj[r] * sdi[jl + 1] * s[jl + 1][r]);
      *(unsigned*)(O + (size_t)(tj * TILE + r) * Nn + (ti * TILE + jl)) = pkd;
    }
  }
}

// ------------------------------ WMMA GEMM ----------------------------------
// D[b] = op( A[b](1024 x K) @ B[b](K x F) + bias ); 64 cols per blockIdx.z.
//   AF32     : A is fp32 row-major (converted on the fly), else bf16 row-major
//   BF32     : B is fp32 row-major [K,F] (register-transposed into LDS);
//              else B is bf16 pre-transposed [F,K] -> straight (async) copy
//   STORET   : store D K-major [F,1024] (feeds next GEMM's B panel)
//   POOL     : column-mean-reduce into pool[b][col] instead of storing D
template<bool AF32, bool BF32, bool BIASRELU, bool POOL, bool STORET>
__global__ __launch_bounds__(256)
void gemm_wmma_kernel(const void* __restrict__ Aptr, const void* __restrict__ Bptr,
                      const float* __restrict__ bias,
                      unsigned short* __restrict__ Dptr, float* __restrict__ pool,
                      int K, int F, int lda,
                      long long strideA, long long strideB, long long strideD,
                      int poolLd, float poolScale) {
  extern __shared__ unsigned short Bt[];   // [64][K] bf16, K-major panel
  const int tid  = threadIdx.x;
  const int lane = tid & 31;
  const int wave = tid >> 5;
  const int half = lane >> 4;
  const int r    = lane & 15;
  const int b    = blockIdx.y;
  const int c0   = blockIdx.z * 64;

  // ---------------- stage 64-column B panel (K-major) into LDS -------------
  if (BF32) {
    const float* Bs = (const float*)Bptr + (long long)b * strideB;
    // pair of K values per thread -> packed dword store to LDS
    for (int i = tid; i < (K >> 1) * 64; i += 256) {
      int k = (i >> 6) * 2, f = i & 63;
      unsigned pkd = pk2bf(Bs[(long long)k * F + c0 + f],
                           Bs[(long long)(k + 1) * F + c0 + f]);
      *(unsigned*)(Bt + f * K + k) = pkd;
    }
    __syncthreads();
  } else {
    // source already [F][K]: contiguous 64*K*2-byte copy
    const unsigned short* Bs = (const unsigned short*)Bptr + (long long)b * strideB
                               + (long long)c0 * K;
    const size_t total = (size_t)64 * K * 2;
#if USE_ASYNC_STAGE
    const unsigned ldsBase = (unsigned)(unsigned long long)(uintptr_t)Bt;
    for (size_t byte = (size_t)tid * 16; byte < total; byte += 256 * 16) {
      unsigned lds = ldsBase + (unsigned)byte;
      unsigned long long g = (unsigned long long)(uintptr_t)((const char*)Bs + byte);
      asm volatile("global_load_async_to_lds_b128 %0, %1, off"
                   :: "v"(lds), "v"(g) : "memory");
    }
    asm volatile("s_wait_asynccnt 0x0" ::: "memory");
    __syncthreads();
#else
    const uint4* src = (const uint4*)Bs;
    uint4* dst = (uint4*)Bt;
    for (size_t i = tid; i < total / 16; i += 256) dst[i] = src[i];
    __syncthreads();
#endif
  }

  const int row0 = (blockIdx.x * 8 + wave) * 16;
  const int rowA = row0 + r;

  const float*          Af = AF32 ? ((const float*)Aptr + (long long)b * strideA + (long long)rowA * lda) : nullptr;
  const unsigned short* Ah = AF32 ? nullptr : ((const unsigned short*)Aptr + (long long)b * strideA + (long long)rowA * lda);

  Frag32 acc[4];
  #pragma unroll
  for (int ft = 0; ft < 4; ++ft)
    #pragma unroll
    for (int g = 0; g < 8; ++g) acc[ft].f[g] = 0.f;

  // Fragment loaders: A = two 16B chunks per lane; B = two ds_load_b128 / tile
  auto loadA = [&](int k0, Frag16& a) {
    if (AF32) {
      const float4* p0 = (const float4*)(Af + k0 + half * 8);
      const float4* p1 = (const float4*)(Af + k0 + 16 + half * 8);
      float4 f0 = p0[0], f1 = p0[1], f2 = p1[0], f3 = p1[1];
      a.u[0] = pk2bf(f0.x, f0.y); a.u[1] = pk2bf(f0.z, f0.w);
      a.u[2] = pk2bf(f1.x, f1.y); a.u[3] = pk2bf(f1.z, f1.w);
      a.u[4] = pk2bf(f2.x, f2.y); a.u[5] = pk2bf(f2.z, f2.w);
      a.u[6] = pk2bf(f3.x, f3.y); a.u[7] = pk2bf(f3.z, f3.w);
    } else {
      a.q[0] = *(const uint4*)(Ah + k0 + half * 8);
      a.q[1] = *(const uint4*)(Ah + k0 + 16 + half * 8);
    }
  };
  auto loadB = [&](int k0, Frag16* bbf) {
    #pragma unroll
    for (int ft = 0; ft < 4; ++ft) {
      const unsigned short* p = Bt + (ft * 16 + r) * K + k0 + half * 8;
      bbf[ft].q[0] = *(const uint4*)(p);
      bbf[ft].q[1] = *(const uint4*)(p + 16);
    }
  };
  auto domma = [&](Frag16& a, Frag16* bbf) {
    #pragma unroll
    for (int ft = 0; ft < 4; ++ft)
      acc[ft].v = __builtin_amdgcn_wmma_f32_16x16x32_bf16(
          false, a.v, false, bbf[ft].v, (short)0, acc[ft].v, false, false);
  };

  // double-buffered K loop (K is a multiple of 64)
  Frag16 a0, a1, b0[4], b1[4];
  loadA(0, a0); loadB(0, b0);
  for (int k0 = 0; k0 < K; k0 += 64) {
    if (k0 + 64 < K) {
      if (AF32) __builtin_prefetch(Af + k0 + 64, 0, 0);
      else      __builtin_prefetch(Ah + k0 + 64, 0, 0);
    }
    if (k0 + 32 < K) { loadA(k0 + 32, a1); loadB(k0 + 32, b1); }
    domma(a0, b0);
    if (k0 + 64 < K) { loadA(k0 + 64, a0); loadB(k0 + 64, b0); }
    if (k0 + 32 < K) domma(a1, b1);
  }

  // ------------------------------ epilogue ---------------------------------
  #pragma unroll
  for (int ft = 0; ft < 4; ++ft) {
    int col = c0 + ft * 16 + r;
    float bv = BIASRELU ? bias[col] : 0.f;
    if (POOL) {
      float s = 0.f;
      #pragma unroll
      for (int g = 0; g < 8; ++g) {
        float v = acc[ft].f[g] + bv;
        if (BIASRELU) v = fmaxf(v, 0.f);
        s += v;
      }
      atomicAdd(pool + (long long)b * poolLd + col, s * poolScale);
    } else if (STORET) {
      // K-major out: rows contiguous along g -> packed dword stores
      #pragma unroll
      for (int g = 0; g < 8; g += 2) {
        float v0 = acc[ft].f[g]     + bv;
        float v1 = acc[ft].f[g + 1] + bv;
        if (BIASRELU) { v0 = fmaxf(v0, 0.f); v1 = fmaxf(v1, 0.f); }
        int row = row0 + g + 8 * half;         // C/D layout: M = g + 8*half
        *(unsigned*)(Dptr + (long long)b * strideD + (long long)col * Nn + row) =
            pk2bf(v0, v1);
      }
    } else {
      #pragma unroll
      for (int g = 0; g < 8; ++g) {
        float v = acc[ft].f[g] + bv;
        if (BIASRELU) v = fmaxf(v, 0.f);
        int row = row0 + g + 8 * half;
        Dptr[(long long)b * strideD + (long long)row * F + col] = f2bf(v);
      }
    }
  }
}

// ------------------------------- head MLP ----------------------------------
__global__ __launch_bounds__(128)
void head_kernel(const float* __restrict__ feat,
                 const float* __restrict__ W1, const float* __restrict__ b1,
                 const float* __restrict__ W2, const float* __restrict__ b2,
                 float* __restrict__ out) {
  __shared__ float h[128];
  const int b = blockIdx.x, t = threadIdx.x;
  float s = b1[t];
  for (int k = 0; k < 256; ++k) s += feat[b * 256 + k] * W1[k * 128 + t];
  h[t] = fmaxf(s, 0.f);
  __syncthreads();
  if (t < 2) {
    float o = b2[t];
    for (int k = 0; k < 128; ++k) o += h[k] * W2[k * 2 + t];
    out[b * 2 + t] = o;
  }
}

// ---------------------------------------------------------------------------
extern "C" void kernel_launch(void* const* d_in, const int* in_sizes, int n_in,
                              void* d_out, int out_size, void* d_ws, size_t ws_size,
                              hipStream_t stream) {
  (void)in_sizes; (void)n_in; (void)out_size; (void)ws_size;
  char* ws = (char*)d_ws;
  size_t off = 0;
  auto alloc = [&](size_t bytes) -> char* {
    char* p = ws + off;
    off = (off + bytes + 255) & ~(size_t)255;
    return p;
  };
  float*          deg  = (float*)alloc((size_t)Bn * Nn * 4);
  unsigned short* Abf  = (unsigned short*)alloc((size_t)Bn * Nn * Nn * 2);   // 128 MB
  unsigned short* C1t  = (unsigned short*)alloc((size_t)Bn * HIDn * Nn * 2); // [b][64][1024]
  unsigned short* H1   = (unsigned short*)alloc((size_t)Bn * Nn * HIDn * 2); // row-major
  unsigned short* C2t  = (unsigned short*)alloc((size_t)Bn * EMBn * Nn * 2); // [b][128][1024]
  float*          emb  = (float*)alloc((size_t)Bn * 2 * EMBn * 4);

  {
    int n = Bn * 2 * EMBn;
    set_kernel<<<(n + 255) / 256, 256, 0, stream>>>(emb, n, 0.f);
  }

  const size_t ldsBig   = (size_t)Nn * 64 * 2;    // 128 KB panel (K=1024)
  const size_t ldsSmall = (size_t)HIDn * 64 * 2;  //   8 KB panel (K=64)

  for (int s = 0; s < 2; ++s) {
    const float* adj = (const float*)d_in[s];
    const int base = 2 + 4 * s;
    const float* W1 = (const float*)d_in[base + 0];
    const float* b1 = (const float*)d_in[base + 1];
    const float* W2 = (const float*)d_in[base + 2];
    const float* b2 = (const float*)d_in[base + 3];

    // --- degree / normalization ---
    {
      int n = Bn * Nn;
      set_kernel<<<(n + 255) / 256, 256, 0, stream>>>(deg, n, 1.0f);  // self loops
      deg_accum_kernel<<<dim3(NPAIRS, Bn), 256, 0, stream>>>(adj, deg);
      finalize_dinv_kernel<<<(n + 255) / 256, 256, 0, stream>>>(deg, n);
      build_A_kernel<<<dim3(NPAIRS, Bn), 256, 0, stream>>>(adj, deg, Abf);
    }

    // --- C1t = (adj @ W1)^T  (fp32 A, fp32 B -> bf16 K-major out) ---
    gemm_wmma_kernel<true, true, false, false, true><<<dim3(8, Bn, 1), 256, ldsBig, stream>>>(
        adj, W1, nullptr, C1t, nullptr,
        Nn, HIDn, Nn, (long long)Nn * Nn, 0, (long long)HIDn * Nn, 0, 0.f);

    // --- H1 = relu(A_norm @ C1 + b1)  (B pre-transposed -> async stage) ---
    gemm_wmma_kernel<false, false, true, false, false><<<dim3(8, Bn, 1), 256, ldsBig, stream>>>(
        Abf, C1t, b1, H1, nullptr,
        Nn, HIDn, Nn, (long long)Nn * Nn, (long long)HIDn * Nn,
        (long long)Nn * HIDn, 0, 0.f);

    // --- C2t = (H1 @ W2)^T  (K=64, F=128 via 2 column groups) ---
    gemm_wmma_kernel<false, true, false, false, true><<<dim3(8, Bn, 2), 256, ldsSmall, stream>>>(
        H1, W2, nullptr, C2t, nullptr,
        HIDn, EMBn, HIDn, (long long)Nn * HIDn, 0, (long long)EMBn * Nn, 0, 0.f);

    // --- emb[:, s*128:] = mean_rows(relu(A_norm @ C2 + b2)) ---
    gemm_wmma_kernel<false, false, true, true, false><<<dim3(8, Bn, 2), 256, ldsBig, stream>>>(
        Abf, C2t, b2, nullptr, emb + s * EMBn,
        Nn, EMBn, Nn, (long long)Nn * Nn, (long long)EMBn * Nn,
        0, 2 * EMBn, 1.0f / (float)Nn);
  }

  head_kernel<<<Bn, 128, 0, stream>>>(
      emb, (const float*)d_in[10], (const float*)d_in[11],
      (const float*)d_in[12], (const float*)d_in[13], (float*)d_out);
}
